// pointnet2_34041910788498
// MI455X (gfx1250) — compile-verified, matrix operand order checked by reading, not run on hardware
//
#include <hip/hip_runtime.h>
#include <hip/hip_bf16.h>

typedef __attribute__((ext_vector_type(16))) _Float16 v16h;
typedef __attribute__((ext_vector_type(8)))  float    v8f;

#define EPSV 1e-5f

// ---------------------------------------------------------------- utilities
static inline int ru16(int x) { return (x + 15) & ~15; }
static inline int ru32(int x) { return (x + 31) & ~31; }

// gfx1250 async global->LDS copy of 16 bytes (tracked by ASYNCcnt)
__device__ __forceinline__ void async_copy_b128(unsigned lds_off, const void* gsrc) {
    unsigned long long ga = (unsigned long long)gsrc;
    asm volatile("global_load_async_to_lds_b128 %0, %1, off"
                 :: "v"(lds_off), "v"(ga) : "memory");
}
__device__ __forceinline__ void wait_async0() {
    asm volatile("s_wait_asynccnt 0x0" ::: "memory");
}

// -------------------------------------------------- (B,3,N) -> (B,N,3)
__global__ void k_tin(const float* __restrict__ x, float* __restrict__ y,
                      int N, int total) {
    int t = blockIdx.x * blockDim.x + threadIdx.x;
    if (t >= total) return;                 // t indexes (b,c,n)
    int n = t % N; int c = (t / N) % 3; int b = t / (3 * N);
    y[((size_t)b * N + n) * 3 + c] = x[t];
}

// -------------------------------------------------- (B,S,C) -> (B,C,S)
__global__ void k_tout(const float* __restrict__ f, float* __restrict__ o,
                       int S, int C, int total) {
    int t = blockIdx.x * blockDim.x + threadIdx.x;
    if (t >= total) return;                 // t indexes (b,c,s)
    int s = t % S; int c = (t / S) % C; int b = t / (S * C);
    o[t] = f[((size_t)b * S + s) * C + c];
}

// -------------------------------------------------- farthest point sampling
__global__ void k_fps(const float* __restrict__ xyz, float* __restrict__ dist,
                      int* __restrict__ fidx, int N, int npoint) {
    const int T = 256;
    int b = blockIdx.x, tid = threadIdx.x;
    __shared__ float sval[T];
    __shared__ int   sidx[T];
    __shared__ int   sfar;
    const float* px = xyz + (size_t)b * N * 3;
    float* pd = dist + (size_t)b * N;
    for (int n = tid; n < N; n += T) pd[n] = 1e10f;
    if (tid == 0) sfar = 0;
    __syncthreads();
    for (int it = 0; it < npoint; ++it) {
        int far = sfar;
        if (tid == 0) fidx[b * npoint + it] = far;
        float cx = px[far * 3 + 0], cy = px[far * 3 + 1], cz = px[far * 3 + 2];
        float bv = -1.0f; int bi = 0x7fffffff;
        for (int n = tid; n < N; n += T) {
            float dx = px[n * 3 + 0] - cx;
            float dy = px[n * 3 + 1] - cy;
            float dz = px[n * 3 + 2] - cz;
            float d  = dx * dx + dy * dy + dz * dz;
            float nd = fminf(pd[n], d);
            pd[n] = nd;
            if (nd > bv || (nd == bv && n < bi)) { bv = nd; bi = n; }
        }
        sval[tid] = bv; sidx[tid] = bi;
        __syncthreads();
        for (int s = T / 2; s > 0; s >>= 1) {
            if (tid < s) {
                float ov = sval[tid + s]; int oi = sidx[tid + s];
                if (ov > sval[tid] || (ov == sval[tid] && oi < sidx[tid])) {
                    sval[tid] = ov; sidx[tid] = oi;
                }
            }
            __syncthreads();
        }
        if (tid == 0) sfar = sidx[0];
        __syncthreads();
    }
}

// -------------------------------------------------- gather sampled centers
__global__ void k_gather_xyz(const float* __restrict__ xyz, const int* __restrict__ fidx,
                             float* __restrict__ nxyz, int N, int S, int total) {
    int t = blockIdx.x * blockDim.x + threadIdx.x;
    if (t >= total) return;                 // t = b*S+s
    int b = t / S;
    int id = fidx[t];
    nxyz[t * 3 + 0] = xyz[((size_t)b * N + id) * 3 + 0];
    nxyz[t * 3 + 1] = xyz[((size_t)b * N + id) * 3 + 1];
    nxyz[t * 3 + 2] = xyz[((size_t)b * N + id) * 3 + 2];
}

// -------------------------------------------------- ball query (first K in index order)
__global__ void k_ball(const float* __restrict__ xyz, const float* __restrict__ nxyz,
                       int* __restrict__ idx, int N, int S, int K, float r2, int total) {
    int t = blockIdx.x * blockDim.x + threadIdx.x;
    if (t >= total) return;                 // t = b*S+s
    int b = t / S;
    const float* px = xyz + (size_t)b * N * 3;
    float cx = nxyz[t * 3 + 0], cy = nxyz[t * 3 + 1], cz = nxyz[t * 3 + 2];
    int* pi = idx + (size_t)t * K;
    int cnt = 0;
    for (int n = 0; n < N; ++n) {
        float dx = px[n * 3 + 0] - cx;
        float dy = px[n * 3 + 1] - cy;
        float dz = px[n * 3 + 2] - cz;
        float d2 = dx * dx + dy * dy + dz * dz;
        if (d2 <= r2) { pi[cnt++] = n; if (cnt >= K) break; }
    }
    int first = (cnt > 0) ? pi[0] : (N - 1);
    for (int j = cnt; j < K; ++j) pi[j] = first;
}

// -------------------------------------------------- group + pack features -> f16 A (M x Kp)
__global__ void k_group(const float* __restrict__ points, int Cp,
                        const float* __restrict__ xyz, const float* __restrict__ nxyz,
                        const int* __restrict__ idx, int N, int S, int K, int Kp,
                        _Float16* __restrict__ A, long total) {
    long t = (long)blockIdx.x * blockDim.x + threadIdx.x;
    if (t >= total) return;
    int  c  = (int)(t % Kp);
    long r  = t / Kp;
    int  k  = (int)(r % K);
    long bs = r / K;                        // b*S+s
    int  b  = (int)(bs / S);
    int id = idx[bs * K + k]; if (id >= N) id = N - 1;
    float val = 0.f;
    if (c < Cp) {
        val = points[((size_t)b * N + id) * Cp + c];
    } else if (c < Cp + 3) {
        int j = c - Cp;
        val = xyz[((size_t)b * N + id) * 3 + j] - nxyz[bs * 3 + j];
    }
    A[(size_t)r * Kp + c] = (_Float16)val;
}

// -------------------------------------------------- sa3 grouping: concat([l2_xyz, l2])
__global__ void k_group3(const float* __restrict__ l2xyz, const float* __restrict__ l2f,
                         _Float16* __restrict__ A, int Kp, int total) {
    int t = blockIdx.x * blockDim.x + threadIdx.x;
    if (t >= total) return;
    int c = t % Kp; int r = t / Kp;         // r = b*128 + j
    float val = 0.f;
    if (c < 3)        val = l2xyz[r * 3 + c];
    else if (c < 515) val = l2f[(size_t)r * 512 + (c - 3)];
    A[(size_t)r * Kp + c] = (_Float16)val;
}

// -------------------------------------------------- pack weights f32(cout,cin) -> f16(Npad,Kp)
__global__ void k_pack_w(const float* __restrict__ W, _Float16* __restrict__ Wp,
                         int cout, int cin, int Kp, int total) {
    int t = blockIdx.x * blockDim.x + threadIdx.x;
    if (t >= total) return;
    int k = t % Kp, n = t / Kp;
    float v = (n < cout && k < cin) ? W[n * cin + k] : 0.f;
    Wp[t] = (_Float16)v;
}

// -------------------------------------------------- WMMA GEMM, async LDS staging, double-buffered
// C(MxNpad) = A(MxKp) * W(NpadxKp)^T + bias
// Block: 256 threads = 8 waves in a 4(M) x 2(N) grid -> 64x32 macro-tile.
// K-chunks of 32 staged into double-buffered LDS via global_load_async_to_lds_b128
// (ASYNCcnt); prefetch of chunk i+1 overlaps the WMMA on chunk i.
__global__ void k_gemm_wmma(const _Float16* __restrict__ A, const _Float16* __restrict__ W,
                            const float* __restrict__ bias, float* __restrict__ C,
                            int M, int Kp, int Npad, int cout) {
    __shared__ _Float16 As[2][64 * 32];   // 2 x 4 KB
    __shared__ _Float16 Bs[2][32 * 32];   // 2 x 2 KB

    int tid  = threadIdx.x;
    int lane = tid & 31;
    int wave = tid >> 5;
    int mw   = wave & 3;                // wave tile row (0..3)
    int nw   = wave >> 2;               // wave tile col (0..1)
    int half = lane >> 4;
    int rc   = lane & 15;

    int rowtile = blockIdx.y * 64;      // macro-tile origin in M
    int coltile = blockIdx.x * 32;      // macro-tile origin in N

    // async staging assignments (16B segments)
    int arow = tid >> 2, aseg = tid & 3;            // 64 rows x 4 segs = 256
    int brow = (tid & 127) >> 2, bseg = tid & 3;    // 32 rows x 4 segs = 128
    unsigned lds_a0 = (unsigned)(unsigned long long)&As[0][arow * 32 + aseg * 8];
    unsigned lds_a1 = (unsigned)(unsigned long long)&As[1][arow * 32 + aseg * 8];
    unsigned lds_b0 = (unsigned)(unsigned long long)&Bs[0][brow * 32 + bseg * 8];
    unsigned lds_b1 = (unsigned)(unsigned long long)&Bs[1][brow * 32 + bseg * 8];
    const _Float16* ga = A + (size_t)(rowtile + arow) * Kp + aseg * 8;
    const _Float16* gb = W + (size_t)(coltile + brow) * Kp + bseg * 8;
    bool doB = (tid < 128);

    const int aoff = (mw * 16 + rc) * 32;
    const int boff = (nw * 16 + rc) * 32;

    int nchunks = Kp >> 5;
    // prologue: stage chunk 0 into buffer 0
    async_copy_b128(lds_a0, ga);
    if (doB) async_copy_b128(lds_b0, gb);

    v8f acc = {};
    for (int i = 0; i < nchunks; ++i) {
        wait_async0();                  // chunk i landed (in flight during prev compute)
        __syncthreads();                // visible to all; prev reads of other buffer done

        if (i + 1 < nchunks) {          // prefetch chunk i+1 into the other buffer
            if ((i + 1) & 1) {
                async_copy_b128(lds_a1, ga + (size_t)(i + 1) * 32);
                if (doB) async_copy_b128(lds_b1, gb + (size_t)(i + 1) * 32);
            } else {
                async_copy_b128(lds_a0, ga + (size_t)(i + 1) * 32);
                if (doB) async_copy_b128(lds_b0, gb + (size_t)(i + 1) * 32);
            }
        }

        const _Float16* ap = &As[i & 1][aoff];
        const _Float16* bp = &Bs[i & 1][boff];
        v16h a, b;
#pragma unroll
        for (int j = 0; j < 8; ++j) {
            a[j]     = ap[half * 8 + j];        // elements k = half*8 .. +7
            a[8 + j] = ap[16 + half * 8 + j];   // elements k = 16+half*8 .. +7
            b[j]     = bp[half * 8 + j];
            b[8 + j] = bp[16 + half * 8 + j];
        }
        acc = __builtin_amdgcn_wmma_f32_16x16x32_f16(false, a, false, b,
                                                     (short)0, acc, false, false);
    }

#pragma unroll
    for (int j = 0; j < 8; ++j) {       // D: row = j + 8*half, col = rc
        int gr = rowtile + mw * 16 + half * 8 + j;
        int gc = coltile + nw * 16 + rc;
        if (gc < cout) C[(size_t)gr * Npad + gc] = acc[j] + bias[gc];
    }
}

// -------------------------------------------------- per-column mean/var over M rows
__global__ void k_stats(const float* __restrict__ C, int M, int Npad,
                        float* __restrict__ mean, float* __restrict__ var) {
    const int T = 256;
    int c = blockIdx.x, tid = threadIdx.x;
    __shared__ float ss[T], sq[T];
    float s = 0.f, q = 0.f;
    for (int r = tid; r < M; r += T) {
        float v = C[(size_t)r * Npad + c];
        s += v; q += v * v;
    }
    ss[tid] = s; sq[tid] = q;
    __syncthreads();
    for (int st = T / 2; st > 0; st >>= 1) {
        if (tid < st) { ss[tid] += ss[tid + st]; sq[tid] += sq[tid + st]; }
        __syncthreads();
    }
    if (tid == 0) {
        float m = ss[0] / (float)M;
        mean[c] = m;
        var[c]  = sq[0] / (float)M - m * m;
    }
}

// -------------------------------------------------- BN + ReLU in place
__global__ void k_bn_relu(float* __restrict__ C, int Npad, int cout,
                          const float* __restrict__ mean, const float* __restrict__ var,
                          const float* __restrict__ g, const float* __restrict__ be,
                          long total) {
    long t = (long)blockIdx.x * blockDim.x + threadIdx.x;
    if (t >= total) return;
    int  c = (int)(t % cout);
    long r = t / cout;
    float* p = C + (size_t)r * Npad + c;
    float v = (*p - mean[c]) * rsqrtf(var[c] + EPSV) * g[c] + be[c];
    *p = v > 0.f ? v : 0.f;
}

// -------------------------------------------------- f32 activations -> padded f16 A
__global__ void k_pack_a(const float* __restrict__ C, _Float16* __restrict__ A,
                         int Npad, int cout, int Kp2, long total) {
    long t = (long)blockIdx.x * blockDim.x + threadIdx.x;
    if (t >= total) return;
    int  c = (int)(t % Kp2);
    long r = t / Kp2;
    float v = (c < cout) ? C[(size_t)r * Npad + c] : 0.f;
    A[(size_t)r * Kp2 + c] = (_Float16)v;
}

// -------------------------------------------------- max over K, scatter into concat buffer
__global__ void k_maxpool(const float* __restrict__ C, float* __restrict__ out,
                          int BS, int K, int Npad, int cout, int Cdst, int coff, int total) {
    int t = blockIdx.x * blockDim.x + threadIdx.x;
    if (t >= total) return;                 // t indexes (p, c)
    int c = t % cout; int p = t / cout;
    float m = -3.4e38f;
    for (int k = 0; k < K; ++k)
        m = fmaxf(m, C[((size_t)p * K + k) * Npad + c]);
    out[(size_t)p * Cdst + coff + c] = m;
}

// ================================================================ host driver
extern "C" void kernel_launch(void* const* d_in, const int* in_sizes, int n_in,
                              void* d_out, int out_size, void* d_ws, size_t ws_size,
                              hipStream_t stream) {
    const int B = 8, N1 = 8192;
    const float* xyz_cf = (const float*)d_in[0];

    struct Lay { const float *W, *b, *g, *be; int cin, cout; };
    int cur = 1;
    auto take = [&](int cin, int cout) {
        Lay L;
        L.W  = (const float*)d_in[cur + 0];
        L.b  = (const float*)d_in[cur + 1];
        L.g  = (const float*)d_in[cur + 2];
        L.be = (const float*)d_in[cur + 3];
        cur += 4; L.cin = cin; L.cout = cout;
        return L;
    };
    int d1[3][4] = {{6,32,32,64},{6,64,64,128},{6,64,96,128}};
    int d2[2][4] = {{323,128,128,256},{323,128,196,256}};
    int d3[4]    = {515,256,512,1024};
    Lay sa1[3][3], sa2[2][3], sa3[3];
    for (int br = 0; br < 3; ++br) for (int l = 0; l < 3; ++l) sa1[br][l] = take(d1[br][l], d1[br][l+1]);
    for (int br = 0; br < 2; ++br) for (int l = 0; l < 3; ++l) sa2[br][l] = take(d2[br][l], d2[br][l+1]);
    for (int l = 0; l < 3; ++l) sa3[l] = take(d3[l], d3[l+1]);

    // ----- workspace carve
    char* ws = (char*)d_ws;
    size_t off = 0;
    auto carve = [&](size_t bytes) -> char* {
        char* p = ws + off; off += (bytes + 255) & ~(size_t)255; return p;
    };
    float*     xyzb  = (float*)carve((size_t)B * N1 * 3 * 4);
    float*     dist  = (float*)carve((size_t)B * N1 * 4);
    int*       fidx1 = (int*)  carve((size_t)B * 512 * 4);
    float*     nx1   = (float*)carve((size_t)B * 512 * 3 * 4);
    float*     l1f   = (float*)carve((size_t)B * 512 * 320 * 4);
    int*       fidx2 = (int*)  carve((size_t)B * 128 * 4);
    float*     nx2   = (float*)carve((size_t)B * 128 * 3 * 4);
    float*     l2f   = (float*)carve((size_t)B * 128 * 512 * 4);
    float*     meanb = (float*)carve(1024 * 4);
    float*     varb  = (float*)carve(1024 * 4);
    int*       idxb  = (int*)  carve((size_t)B * 512 * 128 * 4);
    _Float16*  Wp    = (_Float16*)carve((size_t)1024 * 544 * 2);
    _Float16*  Abuf  = (_Float16*)carve((size_t)524288 * 96 * 2);
    float*     Cbuf  = (float*)carve((size_t)524288 * 128 * 4);

    const int T = 256;
    auto nb = [&](long tot) { return (unsigned)((tot + T - 1) / T); };

    // ----- MLP branch runner
    auto run_mlp = [&](const Lay* Ls, int nl, long M, int Kp0,
                       float* dstFeat, int Cdst, int coff, int BS, int Kk) {
        int Kp = Kp0;
        for (int l = 0; l < nl; ++l) {
            const Lay& L = Ls[l];
            int Np = ru32(L.cout);          // N padded to 32 for the 64x32 macro-tile
            { int tot = Np * Kp;
              k_pack_w<<<nb(tot), T, 0, stream>>>(L.W, Wp, L.cout, L.cin, Kp, tot); }
            { dim3 g((unsigned)(Np / 32), (unsigned)(M / 64));
              k_gemm_wmma<<<g, 256, 0, stream>>>(Abuf, Wp, L.b, Cbuf, (int)M, Kp, Np, L.cout); }
            k_stats<<<L.cout, T, 0, stream>>>(Cbuf, (int)M, Np, meanb, varb);
            { long tot = M * (long)L.cout;
              k_bn_relu<<<nb(tot), T, 0, stream>>>(Cbuf, Np, L.cout, meanb, varb, L.g, L.be, tot); }
            if (l + 1 < nl) {
                int Kp2 = ru32(L.cout);
                long tot = M * (long)Kp2;
                k_pack_a<<<nb(tot), T, 0, stream>>>(Cbuf, Abuf, Np, L.cout, Kp2, tot);
                Kp = Kp2;
            } else {
                int tot = BS * L.cout;
                k_maxpool<<<nb(tot), T, 0, stream>>>(Cbuf, dstFeat, BS, Kk, Np, L.cout, Cdst, coff, tot);
            }
        }
    };

    // ----- input transpose (B,3,N) -> (B,N,3)
    { int tot = B * 3 * N1; k_tin<<<nb(tot), T, 0, stream>>>(xyz_cf, xyzb, N1, tot); }

    // ================= SA1 (S=512, radii {0.1,0.2,0.4}, K {32,64,128})
    k_fps<<<B, T, 0, stream>>>(xyzb, dist, fidx1, N1, 512);
    { int tot = B * 512; k_gather_xyz<<<nb(tot), T, 0, stream>>>(xyzb, fidx1, nx1, N1, 512, tot); }
    {
        float radii[3] = {0.1f, 0.2f, 0.4f};
        int   ks[3]    = {32, 64, 128};
        int   coffs[3] = {0, 64, 192};
        for (int br = 0; br < 3; ++br) {
            int K = ks[br];
            { int tot = B * 512;
              k_ball<<<nb(tot), T, 0, stream>>>(xyzb, nx1, idxb, N1, 512, K,
                                                radii[br] * radii[br], tot); }
            long M = (long)B * 512 * K;
            { long tot = M * 32;   // cin=6 -> Kp=32
              k_group<<<nb(tot), T, 0, stream>>>(xyzb, 3, xyzb, nx1, idxb,
                                                 N1, 512, K, 32, Abuf, tot); }
            run_mlp(sa1[br], 3, M, 32, l1f, 320, coffs[br], B * 512, K);
        }
    }

    // ================= SA2 (points=l1f N=512, S=128, radii {0.4,0.8}, K {64,128})
    k_fps<<<B, T, 0, stream>>>(nx1, dist, fidx2, 512, 128);
    { int tot = B * 128; k_gather_xyz<<<nb(tot), T, 0, stream>>>(nx1, fidx2, nx2, 512, 128, tot); }
    {
        float radii[2] = {0.4f, 0.8f};
        int   ks[2]    = {64, 128};
        int   coffs[2] = {0, 256};
        for (int br = 0; br < 2; ++br) {
            int K = ks[br];
            { int tot = B * 128;
              k_ball<<<nb(tot), T, 0, stream>>>(nx1, nx2, idxb, 512, 128, K,
                                                radii[br] * radii[br], tot); }
            long M = (long)B * 128 * K;
            { long tot = M * 352;  // cin=323 -> Kp=352
              k_group<<<nb(tot), T, 0, stream>>>(l1f, 320, nx1, nx2, idxb,
                                                 512, 128, K, 352, Abuf, tot); }
            run_mlp(sa2[br], 3, M, 352, l2f, 512, coffs[br], B * 128, K);
        }
    }

    // ================= SA3 (feat = concat([l2_xyz, l2]) -> (B,1,128,515))
    float* out0 = (float*)d_out;                        // l1^T : (B,320,512)
    float* out1 = out0 + (size_t)B * 320 * 512;         // l2^T : (B,512,128)
    float* out2 = out1 + (size_t)B * 512 * 128;         // l3^T : (B,1024,1)
    {
        long M = (long)B * 128;  // S=1, K=128
        { int tot = (int)(M * 544);  // cin=515 -> Kp=544
          k_group3<<<nb(tot), T, 0, stream>>>(nx2, l2f, Abuf, 544, tot); }
        run_mlp(sa3, 3, M, 544, out2, 1024, 0, B, 128); // maxpool writes l3 directly
    }

    // ================= output transposes
    { int tot = B * 320 * 512; k_tout<<<nb(tot), T, 0, stream>>>(l1f, out0, 512, 320, tot); }
    { int tot = B * 512 * 128; k_tout<<<nb(tot), T, 0, stream>>>(l2f, out1, 128, 512, tot); }
}
